// BiLSTM_49435073577828
// MI455X (gfx1250) — matrix operand order repaired and checked by reading, not compiled, not run on hardware
//
#include <hip/hip_runtime.h>
#include <hip/hip_bf16.h>

// ---------------------------------------------------------------------------
// BiLSTM for MI455X (gfx1250): bf16 WMMA (v_wmma_f32_16x16x32_bf16, wave32)
//   Phase 0: convert x and the 8 weight matrices to bf16 (Wx / Wh split).
//   Phase 1: parallel GEMM  Zx[t,d,g] = x_t @ Wx^T  (all timesteps at once).
//            Wave tile = 32(M) x 64(N): 8 WMMAs per 12 b128 loads per k-tile.
//   Phase 2: 256 sequential step kernels:
//            gates = Zx + b + h_{t-1} @ Wh^T ; c,h update ; write output.
// ---------------------------------------------------------------------------

typedef __attribute__((ext_vector_type(16))) __bf16 v16bf;
typedef __attribute__((ext_vector_type(8)))  float  v8f;

#define B_ 64
#define L_ 256
#define D_ 512
#define H_ 512

constexpr int    BH  = B_ * H_;        // 32768
constexpr int    ML  = B_ * L_;        // 16384 rows of flattened x
constexpr int    DH  = D_ + H_;        // 1024

// ---- workspace layout (bytes) ----
constexpr size_t OFF_XB  = 0;                              // ML*D bf16
constexpr size_t SZ_XB   = (size_t)ML * D_ * 2;
constexpr size_t OFF_WXB = OFF_XB + SZ_XB;                 // 8 * H*D bf16
constexpr size_t SZ_WXB  = (size_t)8 * H_ * D_ * 2;
constexpr size_t OFF_WHB = OFF_WXB + SZ_WXB;               // 8 * H*H bf16
constexpr size_t SZ_WHB  = (size_t)8 * H_ * H_ * 2;
constexpr size_t OFF_HB  = OFF_WHB + SZ_WHB;               // 2 bufs * 2 dirs * BH bf16
constexpr size_t SZ_HB   = (size_t)2 * 2 * BH * 2;
constexpr size_t OFF_C   = OFF_HB + SZ_HB;                 // 2 dirs * BH f32
constexpr size_t SZ_C    = (size_t)2 * BH * 4;
constexpr size_t OFF_ZX  = OFF_C + SZ_C;                   // 2*4*ML*H f32 (268 MB)

__device__ __forceinline__ unsigned short f2bf(float f) {
    unsigned int u = __float_as_uint(f);
    unsigned int r = u + 0x7FFFu + ((u >> 16) & 1u);       // round-to-nearest-even
    return (unsigned short)(r >> 16);
}

__device__ __forceinline__ float sigmoidf_(float x) {
    return 1.0f / (1.0f + __expf(-x));
}

struct WPtrs    { const float* w[8]; };
struct BiasPtrs { const float* b[8]; };

// ---------------------------------------------------------------------------
// Converters / init
// ---------------------------------------------------------------------------
__global__ void cvt_x_kernel(const float* __restrict__ x,
                             unsigned short* __restrict__ xb, int n) {
    for (int i = blockIdx.x * blockDim.x + threadIdx.x; i < n;
         i += gridDim.x * blockDim.x)
        xb[i] = f2bf(x[i]);
}

__global__ void cvt_w_kernel(WPtrs wp,
                             unsigned short* __restrict__ wxb,
                             unsigned short* __restrict__ whb) {
    const int per = H_ * DH;                 // elems per weight matrix
    const int total = 8 * per;
    for (int i = blockIdx.x * blockDim.x + threadIdx.x; i < total;
         i += gridDim.x * blockDim.x) {
        int wi  = i / per;
        int r   = i - wi * per;
        int n   = r / DH;
        int col = r - n * DH;
        unsigned short v = f2bf(wp.w[wi][(size_t)n * DH + col]);
        if (col < D_) wxb[(size_t)wi * H_ * D_ + (size_t)n * D_ + col] = v;
        else          whb[(size_t)wi * H_ * H_ + (size_t)n * H_ + (col - D_)] = v;
    }
}

__global__ void init_state_kernel(float* __restrict__ c,
                                  unsigned short* __restrict__ hb) {
    for (int i = blockIdx.x * blockDim.x + threadIdx.x; i < 2 * BH;
         i += gridDim.x * blockDim.x)
        c[i] = 0.0f;
    for (int i = blockIdx.x * blockDim.x + threadIdx.x; i < 2 * 2 * BH;
         i += gridDim.x * blockDim.x)
        hb[i] = 0;                            // bf16 zero
}

// ---------------------------------------------------------------------------
// Phase 1: Zx[t,d,g][b,h] = sum_k x[b,t,k] * Wx[d,g][h,k]
// grid (512, 16), block 128 (4 waves). Wave = one 32(M) x 64(N) tile:
// per k-tile: 2 A frags (4 loads) + 4 B frags (8 loads) -> 8 WMMAs.
// ---------------------------------------------------------------------------
__global__ void xw_gemm_kernel(const unsigned short* __restrict__ xb,
                               const unsigned short* __restrict__ wxb,
                               float* __restrict__ zx) {
    const int lane  = threadIdx.x & 31;
    const int wv    = threadIdx.x >> 5;
    const int mbase = blockIdx.x * 32;                  // 2 M-tiles per wave
    const int ncat0 = blockIdx.y * 256 + wv * 64;       // 0..4095, 64-aligned
    const int d  = ncat0 >> 11;
    const int g  = (ncat0 >> 9) & 3;
    const int h0 = ncat0 & 511;
    const int nlo = lane & 15;
    const int kh  = lane >> 4;

    const unsigned short* wbase = wxb + (size_t)(d * 4 + g) * H_ * D_;
    const unsigned short* arow0 = xb + (size_t)(mbase + nlo) * D_;
    const unsigned short* arow1 = xb + (size_t)(mbase + 16 + nlo) * D_;

    v8f acc[2][4];
    #pragma unroll
    for (int mi = 0; mi < 2; ++mi)
        #pragma unroll
        for (int j = 0; j < 4; ++j)
            acc[mi][j] = (v8f){0.f, 0.f, 0.f, 0.f, 0.f, 0.f, 0.f, 0.f};

    #pragma unroll 2
    for (int kt = 0; kt < D_ / 32; ++kt) {
        const int k0 = kt * 32 + kh * 16;
        union { v16bf v; uint4 q[2]; } a0, a1;
        a0.q[0] = *(const uint4*)(arow0 + k0);          // K = k0 .. k0+7
        a0.q[1] = *(const uint4*)(arow0 + k0 + 8);      // K = k0+8 .. k0+15
        a1.q[0] = *(const uint4*)(arow1 + k0);
        a1.q[1] = *(const uint4*)(arow1 + k0 + 8);
        #pragma unroll
        for (int j = 0; j < 4; ++j) {
            const int n = h0 + j * 16 + nlo;
            const v16bf bfrag =
                *(const v16bf*)(wbase + (size_t)n * D_ + kt * 32 + kh * 16);
            acc[0][j] = __builtin_amdgcn_wmma_f32_16x16x32_bf16(
                false, a0.v, false, bfrag, (short)0, acc[0][j], false, false);
            acc[1][j] = __builtin_amdgcn_wmma_f32_16x16x32_bf16(
                false, a1.v, false, bfrag, (short)0, acc[1][j], false, false);
        }
    }

    // C/D layout: lane l, vgpr v -> (M = v + 8*(l>>4), N = l&15)
    #pragma unroll
    for (int mi = 0; mi < 2; ++mi) {
        #pragma unroll
        for (int j = 0; j < 4; ++j) {
            const int h = h0 + j * 16 + nlo;
            #pragma unroll
            for (int v = 0; v < 8; ++v) {
                const int m = mbase + mi * 16 + v + 8 * kh;
                const int b = m >> 8;                   // L_ == 256
                const int t = m & 255;
                zx[(size_t)((t * 2 + d) * 4 + g) * BH + (size_t)b * H_ + h] =
                    acc[mi][j][v];
            }
        }
    }
}

// ---------------------------------------------------------------------------
// Phase 2: one launch per step s.
// grid (32, 2): x = 16-wide h slice, y = direction. block 128 (4 waves).
// Wave wv owns batch rows 16wv..16wv+15 for ALL 4 gates of its h slice, so
// the cell update (c,h) is lane-local after the k loop.
// ---------------------------------------------------------------------------
__global__ void lstm_step_kernel(const float* __restrict__ zx,
                                 const unsigned short* __restrict__ whb,
                                 unsigned short* __restrict__ hb,
                                 float* __restrict__ c,
                                 float* __restrict__ out,
                                 BiasPtrs bp, int s) {
    const int lane = threadIdx.x & 31;
    const int wv   = threadIdx.x >> 5;        // batch tile
    const int d    = blockIdx.y;
    const int hc   = blockIdx.x * 16;
    const int t    = d ? (L_ - 1 - s) : s;
    const int nlo  = lane & 15;
    const int kh   = lane >> 4;
    const int hidx = hc + nlo;

    // init accumulators with Zx + bias
    v8f acc[4];
    #pragma unroll
    for (int g = 0; g < 4; ++g) {
        const float* zt = zx + (size_t)((t * 2 + d) * 4 + g) * BH;
        const float bias = bp.b[d * 4 + g][hidx];
        #pragma unroll
        for (int v = 0; v < 8; ++v)
            acc[g][v] = zt[(size_t)(wv * 16 + v + 8 * kh) * H_ + hidx] + bias;
    }

    // h_{t-1} @ Wh^T : A from bf16 h state (double-buffered), B from Wh
    const unsigned short* hprev =
        hb + (size_t)((s & 1) * 2 + d) * BH;
    const unsigned short* arow = hprev + (size_t)(wv * 16 + nlo) * H_;

    #pragma unroll 2
    for (int kt = 0; kt < H_ / 32; ++kt) {
        const int k0 = kt * 32 + kh * 16;
        union { v16bf v; uint4 q[2]; } a;
        a.q[0] = *(const uint4*)(arow + k0);
        a.q[1] = *(const uint4*)(arow + k0 + 8);
        #pragma unroll
        for (int g = 0; g < 4; ++g) {
            const unsigned short* wb = whb + (size_t)(d * 4 + g) * H_ * H_;
            const v16bf bfrag =
                *(const v16bf*)(wb + (size_t)hidx * H_ + kt * 32 + kh * 16);
            acc[g] = __builtin_amdgcn_wmma_f32_16x16x32_bf16(
                false, a.v, false, bfrag, (short)0, acc[g], false, false);
        }
    }

    // cell update, lane-local: lane has all 4 gates at (b = wv*16+v+8*kh, hidx)
    float* cd = c + (size_t)d * BH;
    unsigned short* hnext = hb + (size_t)(((s + 1) & 1) * 2 + d) * BH;
    #pragma unroll
    for (int v = 0; v < 8; ++v) {
        const int b = wv * 16 + v + 8 * kh;
        const size_t ci = (size_t)b * H_ + hidx;
        const float ft = sigmoidf_(acc[0][v]);
        const float it = sigmoidf_(acc[1][v]);
        const float ot = sigmoidf_(acc[2][v]);
        const float gt = tanhf(acc[3][v]);
        const float cn = ft * cd[ci] + it * gt;
        cd[ci] = cn;
        const float hn = ot * tanhf(cn);
        out[((size_t)b * L_ + t) * (2 * H_) + (size_t)d * H_ + hidx] = hn;
        hnext[ci] = f2bf(hn);
    }
}

// ---------------------------------------------------------------------------
extern "C" void kernel_launch(void* const* d_in, const int* in_sizes, int n_in,
                              void* d_out, int out_size, void* d_ws,
                              size_t ws_size, hipStream_t stream) {
    (void)in_sizes; (void)n_in; (void)out_size; (void)ws_size;

    const float* x = (const float*)d_in[0];
    WPtrs wp; BiasPtrs bps;
    for (int d = 0; d < 2; ++d)
        for (int g = 0; g < 4; ++g) {
            wp.w[d * 4 + g]  = (const float*)d_in[1 + d * 8 + g * 2];
            bps.b[d * 4 + g] = (const float*)d_in[2 + d * 8 + g * 2];
        }

    char* ws = (char*)d_ws;
    unsigned short* xb  = (unsigned short*)(ws + OFF_XB);
    unsigned short* wxb = (unsigned short*)(ws + OFF_WXB);
    unsigned short* whb = (unsigned short*)(ws + OFF_WHB);
    unsigned short* hb  = (unsigned short*)(ws + OFF_HB);
    float*          c   = (float*)(ws + OFF_C);
    float*          zx  = (float*)(ws + OFF_ZX);
    float*          out = (float*)d_out;

    cvt_x_kernel<<<dim3(2048), dim3(256), 0, stream>>>(x, xb, ML * D_);
    cvt_w_kernel<<<dim3(2048), dim3(256), 0, stream>>>(wp, wxb, whb);
    init_state_kernel<<<dim3(256), dim3(256), 0, stream>>>(c, hb);

    // input projections for all timesteps/gates/dirs at once
    xw_gemm_kernel<<<dim3(ML / 32, 16), dim3(128), 0, stream>>>(xb, wxb, zx);

    // sequential recurrence (fw and bw in parallel via grid.y)
    for (int s = 0; s < L_; ++s)
        lstm_step_kernel<<<dim3(H_ / 16, 2), dim3(128), 0, stream>>>(
            zx, whb, hb, c, out, bps, s);
}
